// Contrastive_Loss_12163347382707
// MI455X (gfx1250) — matrix operand order repaired and checked by reading, not compiled, not run on hardware
//
#include <hip/hip_runtime.h>
#include <hip/hip_bf16.h>

typedef __attribute__((ext_vector_type(16))) _Float16 v16h;
typedef __attribute__((ext_vector_type(8)))  float    v8f;

// Problem dims (fixed by reference setup_inputs)
constexpr int B = 4, C = 64, T = 16, N = 256, H = 128, W = 128;
constexpr int HW   = H * W;        // 16384
constexpr int THW  = T * H * W;    // feature channel stride
constexpr int NPTS = B * T * N;    // 16384 gathered points

// Wave counts per lag l (one wave per 16x16 sim tile, 256 tiles per (b,ta))
constexpr int W0 = B * 16 * 256;           // 16384
constexpr int W1 = W0 + B * 15 * 256;      // 31744
constexpr int W2 = W1 + B * 14 * 256;      // 46080
constexpr int W3 = W2 + B * 13 * 256;      // 59392 total waves

// Workspace layout:
//   [0..31]   u64 se_acc[4]   (Q40.24 fixed-point squared-error sums)
//   [32..47]  u32 cnt_acc[4]
//   [48..51]  float mse_sum (unnormalized)
//   [64.. ]   _Float16 selh[B*T*N*C]  (2 MB staged normalized features)

__global__ void k_mse_init(const float* __restrict__ pred,
                           const float* __restrict__ tgt, void* ws) {
  unsigned long long* se  = (unsigned long long*)ws;
  unsigned int*       cnt = (unsigned int*)((char*)ws + 32);
  float*              msp = (float*)((char*)ws + 48);
  int tid = threadIdx.x;
  if (tid < 4) { se[tid] = 0ull; cnt[tid] = 0u; }
  __shared__ float red[1024];
  float s = 0.f;
  for (int i = tid; i < B * HW; i += 1024) {
    float d = pred[i] - tgt[i];
    s += d * d;
  }
  red[tid] = s;
  __syncthreads();
  for (int off = 512; off > 0; off >>= 1) {
    if (tid < off) red[tid] += red[tid + off];
    __syncthreads();
  }
  if (tid == 0) *msp = red[0];
}

__global__ void k_gather(const float* __restrict__ feat,
                         const int* __restrict__ coord, void* ws) {
  int p = blockIdx.x * blockDim.x + threadIdx.x;  // point id in [0, NPTS)
  if (p >= NPTS) return;
  _Float16* selh = (_Float16*)((char*)ws + 64);
  int x = coord[2 * p + 0];
  int y = coord[2 * p + 1];
  int xc = min(max(x, 0), W - 1);
  int yc = min(max(y, 0), H - 1);
  int b = p / (T * N);
  int t = (p % (T * N)) / N;
  const float* fp = feat + ((size_t)b * C * T + t) * HW + (size_t)yc * W + xc;
  float v[C];
  float ss = 0.f;
#pragma unroll
  for (int c = 0; c < C; ++c) {
    float f = fp[(size_t)c * THW];
    v[c] = f;
    ss += f * f;
  }
  float inv = 1.0f / fmaxf(sqrtf(ss), 1e-12f);
  _Float16* op = selh + (size_t)p * C;
#pragma unroll
  for (int c = 0; c < C; ++c) op[c] = (_Float16)(v[c] * inv);
}

__global__ void k_sim(const int* __restrict__ coord,
                      const int* __restrict__ label, void* ws) {
  unsigned long long* se_acc  = (unsigned long long*)ws;
  unsigned int*       cnt_acc = (unsigned int*)((char*)ws + 32);
  const _Float16*     selh    = (const _Float16*)((const char*)ws + 64);

  int wave   = blockIdx.x * (blockDim.x >> 5) + (threadIdx.x >> 5);
  int lane   = threadIdx.x & 31;
  int laneLo = lane & 15;
  int hi     = lane >> 4;

  int l, rem;
  if      (wave < W0) { l = 0; rem = wave; }
  else if (wave < W1) { l = 1; rem = wave - W0; }
  else if (wave < W2) { l = 2; rem = wave - W1; }
  else                { l = 3; rem = wave - W2; }
  int combo = rem >> 8;        // (b, ta) index
  int tile  = rem & 255;       // 16x16 grid of 16x16 tiles
  int tileM = tile >> 4, tileN = tile & 15;
  int tlen = T - l;
  int b  = combo / tlen;
  int ta = combo % tlen;
  int tb = ta + l;
  int thr2 = (l <= 1) ? 625 : (l == 2 ? 900 : 1225);  // 25^2,25^2,30^2,35^2

  // ---- WMMA gram tile: sim[m,n] = dot(fa row m, fb row n), K = 64 ----
  // Distinct fragment variables so all 8 b128 loads issue in one clause and
  // WMMA #1 only needs a partial s_wait_loadcnt.
  const _Float16* pa = selh + ((size_t)((b * T + ta) * N) + tileM * 16 + laneLo) * C;
  const _Float16* pb = selh + ((size_t)((b * T + tb) * N) + tileN * 16 + laneLo) * C;
  // A (16x32 f16, M x K): lane<16 holds K=k0+[0..15] of its row,
  //                       lane>=16 holds K=k0+[16..31]
  const _Float16* ap0 = pa + 0  + hi * 16;
  const _Float16* ap1 = pa + 32 + hi * 16;
  // B (32x16 f16, K x N): lane<16 holds K=k0+[0..15] of its column,
  //                       lane>=16 holds K=k0+[16..31]
  const _Float16* bp0 = pb + 0  + hi * 16;
  const _Float16* bp1 = pb + 32 + hi * 16;

  v16h a0, a1, b0, b1;
#pragma unroll
  for (int e = 0; e < 8; ++e) {
    a0[e] = ap0[e]; a0[e + 8] = ap0[8 + e];
    a1[e] = ap1[e]; a1[e + 8] = ap1[8 + e];
  }
#pragma unroll
  for (int e = 0; e < 16; ++e) { b0[e] = bp0[e]; b1[e] = bp1[e]; }

  // ---- metadata loads (column point per lane; 8 contiguous rows per half) ----
  int nB   = tileN * 16 + laneLo;
  int idxB = (b * T + tb) * N + nB;
  int cbx = coord[2 * idxB + 0];
  int cby = coord[2 * idxB + 1];
  int lb  = label[idxB];

  int baseA = (b * T + ta) * N + tileM * 16 + hi * 8;  // multiple of 8
  const int4* lp = (const int4*)(label + baseA);        // 32B aligned
  const int4* cp = (const int4*)(coord + 2 * baseA);    // 64B aligned
  int4 lv0 = lp[0], lv1 = lp[1];
  int4 cv0 = cp[0], cv1 = cp[1], cv2 = cp[2], cv3 = cp[3];
  int laArr[8]  = {lv0.x, lv0.y, lv0.z, lv0.w, lv1.x, lv1.y, lv1.z, lv1.w};
  int caxArr[8] = {cv0.x, cv0.z, cv1.x, cv1.z, cv2.x, cv2.z, cv3.x, cv3.z};
  int cayArr[8] = {cv0.y, cv0.w, cv1.y, cv1.w, cv2.y, cv2.w, cv3.y, cv3.w};

  v8f acc = {};
  acc = __builtin_amdgcn_wmma_f32_16x16x32_f16(
      false, a0, false, b0, (short)0, acc, false, false);
  acc = __builtin_amdgcn_wmma_f32_16x16x32_f16(
      false, a1, false, b1, (short)0, acc, false, false);

  // ---- branchless fused masked squared-error ----
  // C/D layout: lane's column n = laneLo; VGPR r holds row m = r + 8*hi
  int notPadB = (cbx >= 0) ? 1 : 0;
  float se = 0.f;
  int   cnt = 0;
#pragma unroll
  for (int r = 0; r < 8; ++r) {
    int dx = caxArr[r] - cbx;
    int dy = cayArr[r] - cby;
    int d2 = dx * dx + dy * dy;
    int ok = (d2 < thr2) & ((caxArr[r] >= 0) ? 1 : 0) & notPadB;
    float lbl = (laArr[r] == lb) ? 1.0f : 0.0f;
    float d = acc[r] - lbl;
    se += (float)ok * d * d;
    cnt += ok;
  }

  // wave32 reduction, then deterministic integer atomics
  for (int off = 16; off > 0; off >>= 1) {
    se  += __shfl_down(se, off, 32);
    cnt += __shfl_down(cnt, off, 32);
  }
  if (lane == 0) {
    unsigned long long fx = (unsigned long long)((double)se * 16777216.0 + 0.5);
    atomicAdd(&se_acc[l], fx);
    atomicAdd(&cnt_acc[l], (unsigned int)cnt);
  }
}

__global__ void k_final(void* ws, float* out) {
  const unsigned long long* se  = (const unsigned long long*)ws;
  const unsigned int*       cnt = (const unsigned int*)((const char*)ws + 32);
  const float*              msp = (const float*)((const char*)ws + 48);
  double loss = (double)(*msp) / (double)(B * HW);
  for (int l = 0; l < 4; ++l) {
    double s = (double)se[l] * (1.0 / 16777216.0);
    double c = (double)(cnt[l] ? cnt[l] : 1u);
    loss += 0.25 * (s / c);
  }
  out[0] = (float)loss;
}

extern "C" void kernel_launch(void* const* d_in, const int* in_sizes, int n_in,
                              void* d_out, int out_size, void* d_ws, size_t ws_size,
                              hipStream_t stream) {
  const float* predicts = (const float*)d_in[0];  // (B,1,H,W) f32
  const float* feature  = (const float*)d_in[1];  // (B,C,T,H,W) f32
  const int*   coord    = (const int*)d_in[2];    // (B,T,N,2) i32
  const float* targets  = (const float*)d_in[3];  // (B,H,W) f32
  const int*   label    = (const int*)d_in[4];    // (B,T,N) i32
  float* out = (float*)d_out;

  k_mse_init<<<1, 1024, 0, stream>>>(predicts, targets, d_ws);
  k_gather<<<NPTS / 256, 256, 0, stream>>>(feature, coord, d_ws);
  k_sim<<<W3 / 8, 256, 0, stream>>>(coord, label, d_ws);  // 8 waves/block
  k_final<<<1, 1, 0, stream>>>(d_ws, out);
}
